// KGCompletionGNN_84963043049955
// MI455X (gfx1250) — compile-verified, matrix operand order checked by reading, not run on hardware
//
#include <hip/hip_runtime.h>
#include <hip/hip_bf16.h>

// ---------------------------------------------------------------------------
// CDNA5 (gfx1250, wave32) implementation of the KG-completion GNN forward.
// All GEMMs run on v_wmma_f32_16x16x32_f16: one wave computes a 16x64 output
// tile (4 accumulators), A fragments are converted fp32->f16 in registers,
// B (weight) fragments are pre-swizzled once into WMMA lane layout.
// Segment-sum is fused into the message GEMM epilogue via global f32 atomics.
// ---------------------------------------------------------------------------

typedef __attribute__((ext_vector_type(16))) _Float16 v16h;
typedef __attribute__((ext_vector_type(8)))  float    v8f;

#define WPB 8            // waves per block
#define BLOCK (WPB * 32)

__device__ __forceinline__ float leakyf(float x) { return x >= 0.f ? x : 0.01f * x; }

// K index inside a 32-wide K tile for fragment element j of `lane`
// (matches ISA 16-bit A 16x32 layout; B uses the mirrored layout with N=lane&15).
__device__ __forceinline__ int frag_k(int lane, int j) {
    int khalf = (lane >> 4) & 1;
    return (j < 8 ? j : j + 8) + khalf * 8;
}

// Load one 16x32 f16 A fragment for this lane. `row` points at the fp32 data
// for this lane's matrix row; k0 is the K-tile base (multiple of 32).
__device__ __forceinline__ v16h load_a_frag(const float* __restrict__ row, int k0, int lane) {
    const int khalf = (lane >> 4) & 1;
    const float4* p0 = (const float4*)(row + k0 + khalf * 8);
    const float4* p1 = (const float4*)(row + k0 + 16 + khalf * 8);
    float4 x0 = p0[0], x1 = p0[1];
    float4 y0 = p1[0], y1 = p1[1];
    v16h a;
    a[0]  = (_Float16)x0.x; a[1]  = (_Float16)x0.y; a[2]  = (_Float16)x0.z; a[3]  = (_Float16)x0.w;
    a[4]  = (_Float16)x1.x; a[5]  = (_Float16)x1.y; a[6]  = (_Float16)x1.z; a[7]  = (_Float16)x1.w;
    a[8]  = (_Float16)y0.x; a[9]  = (_Float16)y0.y; a[10] = (_Float16)y0.z; a[11] = (_Float16)y0.w;
    a[12] = (_Float16)y1.x; a[13] = (_Float16)y1.y; a[14] = (_Float16)y1.z; a[15] = (_Float16)y1.w;
    return a;
}

// Accumulate over NCH chunks of 64 K-values; each chunk has its own row ptr.
template <int NCH>
__device__ __forceinline__ void wmma_chunks(const float* const (&srcs)[NCH],
                                            const v16h* __restrict__ bfrag,
                                            int lane, v8f (&c)[4]) {
#pragma unroll
    for (int kt = 0; kt < NCH * 2; ++kt) {
        v16h a = load_a_frag(srcs[kt >> 1], (kt & 1) * 32, lane);
#pragma unroll
        for (int t = 0; t < 4; ++t) {
            v16h b = bfrag[(kt * 4 + t) * 32 + lane];
            c[t] = __builtin_amdgcn_wmma_f32_16x16x32_f16(false, a, false, b,
                                                          (short)0, c[t], false, false);
        }
    }
}

// Accumulate over a contiguous K (multiple of 32) from one row pointer.
__device__ __forceinline__ void wmma_contig(const float* __restrict__ row, int K,
                                            const v16h* __restrict__ bfrag,
                                            int lane, v8f (&c)[4]) {
    const int nkt = K >> 5;
    for (int kt = 0; kt < nkt; ++kt) {
        v16h a = load_a_frag(row, kt * 32, lane);
#pragma unroll
        for (int t = 0; t < 4; ++t) {
            v16h b = bfrag[(kt * 4 + t) * 32 + lane];
            c[t] = __builtin_amdgcn_wmma_f32_16x16x32_f16(false, a, false, b,
                                                          (short)0, c[t], false, false);
        }
    }
}

// Un-stripe the four 16x16 f32 C tiles into a 16x64 row-major LDS tile.
__device__ __forceinline__ void store_c_lds(float* __restrict__ s, const v8f (&c)[4], int lane) {
    const int col = lane & 15;
    const int rb  = ((lane >> 4) & 1) * 8;
#pragma unroll
    for (int t = 0; t < 4; ++t)
#pragma unroll
        for (int r = 0; r < 8; ++r)
            s[(rb + r) * 64 + t * 16 + col] = c[t][r];
}

// ---------------------------------------------------------------------------
// Weight packing: fp32 [K][64] row-major -> f16 fragments in WMMA lane order.
// ---------------------------------------------------------------------------
__global__ void __launch_bounds__(256) pack_b_kernel(const float* __restrict__ src,
                                                     _Float16* __restrict__ dst, int K) {
    int total = K * 64;
    for (int id = blockIdx.x * blockDim.x + threadIdx.x; id < total;
         id += gridDim.x * blockDim.x) {
        int j    = id & 15;
        int lane = (id >> 4) & 31;
        int tile = id >> 9;
        int kt = tile >> 2, nt = tile & 3;
        int k  = kt * 32 + frag_k(lane, j);
        int n  = nt * 16 + (lane & 15);
        dst[id] = (_Float16)src[k * 64 + n];
    }
}

__global__ void __launch_bounds__(256) zero_kernel(float* __restrict__ p, int n) {
    for (int i = blockIdx.x * blockDim.x + threadIdx.x; i < n; i += gridDim.x * blockDim.x)
        p[i] = 0.f;
}

// ---------------------------------------------------------------------------
// H0 = LN(leaky(entity_feat @ W_ent + b)); also H = H0.
// ---------------------------------------------------------------------------
__global__ void __launch_bounds__(BLOCK) node_proj_kernel(
    const float* __restrict__ feat, const v16h* __restrict__ bfrag,
    const float* __restrict__ bias, const float* __restrict__ g, const float* __restrict__ be,
    float* __restrict__ H0, float* __restrict__ H, int nrows) {
    __shared__ float lds[WPB][16 * 64];
    const int lane = threadIdx.x & 31, wave = threadIdx.x >> 5;
    const int tile = blockIdx.x * WPB + wave;
    if (tile * 16 >= nrows) return;  // wave-uniform
    const float* arow = feat + (size_t)(tile * 16 + (lane & 15)) * 768;
    v8f c[4] = {};
    wmma_contig(arow, 768, bfrag, lane, c);
    float* s = lds[wave];
    store_c_lds(s, c, lane);
    if (lane < 16) {
        float* row = s + lane * 64;
        float mu = 0.f;
        for (int k = 0; k < 64; ++k) { float x = leakyf(row[k] + bias[k]); row[k] = x; mu += x; }
        mu *= (1.f / 64.f);
        float var = 0.f;
        for (int k = 0; k < 64; ++k) { float d = row[k] - mu; var += d * d; }
        float inv = rsqrtf(var * (1.f / 64.f) + 1e-5f);
        size_t o = (size_t)(tile * 16 + lane) * 64;
        for (int k = 0; k < 64; ++k) {
            float y = (row[k] - mu) * inv * g[k] + be[k];
            H0[o + k] = y; H[o + k] = y;
        }
    }
}

// ---------------------------------------------------------------------------
// E = LN(leaky(rel_table[r_tensor] @ W_edge_in + b))   (gathered A rows)
// ---------------------------------------------------------------------------
__global__ void __launch_bounds__(BLOCK) edge_proj_kernel(
    const float* __restrict__ rel_table, const int* __restrict__ r_tensor,
    const v16h* __restrict__ bfrag,
    const float* __restrict__ bias, const float* __restrict__ g, const float* __restrict__ be,
    float* __restrict__ E, int nrows) {
    __shared__ float lds[WPB][16 * 64];
    const int lane = threadIdx.x & 31, wave = threadIdx.x >> 5;
    const int tile = blockIdx.x * WPB + wave;
    if (tile * 16 >= nrows) return;
    const int e = tile * 16 + (lane & 15);
    const float* arow = rel_table + (size_t)r_tensor[e] * 768;
    v8f c[4] = {};
    wmma_contig(arow, 768, bfrag, lane, c);
    float* s = lds[wave];
    store_c_lds(s, c, lane);
    if (lane < 16) {
        float* row = s + lane * 64;
        float mu = 0.f;
        for (int k = 0; k < 64; ++k) { float x = leakyf(row[k] + bias[k]); row[k] = x; mu += x; }
        mu *= (1.f / 64.f);
        float var = 0.f;
        for (int k = 0; k < 64; ++k) { float d = row[k] - mu; var += d * d; }
        float inv = rsqrtf(var * (1.f / 64.f) + 1e-5f);
        size_t o = (size_t)(tile * 16 + lane) * 64;
        for (int k = 0; k < 64; ++k)
            E[o + k] = (row[k] - mu) * inv * g[k] + be[k];
    }
}

// ---------------------------------------------------------------------------
// Query compaction: qi = nonzero(queries)[:NQ], stable order.
// ---------------------------------------------------------------------------
__global__ void __launch_bounds__(256) count_chunks_kernel(const int* __restrict__ q,
                                                           int* __restrict__ ccnt, int M_) {
    __shared__ int wsum[WPB];
    int i = blockIdx.x * 256 + threadIdx.x;
    int v = (i < M_ && q[i] != 0) ? 1 : 0;
    unsigned long long m = __ballot(v);
    if ((threadIdx.x & 31) == 0) wsum[threadIdx.x >> 5] = __popcll(m);
    __syncthreads();
    if (threadIdx.x == 0) {
        int s = 0;
        for (int k = 0; k < WPB; ++k) s += wsum[k];
        ccnt[blockIdx.x] = s;
    }
}

__global__ void scan_chunks_kernel(const int* __restrict__ ccnt, int* __restrict__ coff, int n) {
    if (blockIdx.x == 0 && threadIdx.x == 0) {
        int s = 0;
        for (int i = 0; i < n; ++i) { coff[i] = s; s += ccnt[i]; }
    }
}

__global__ void __launch_bounds__(256) write_qi_kernel(const int* __restrict__ q,
                                                       const int* __restrict__ coff,
                                                       int* __restrict__ qi, int M_, int NQ_) {
    __shared__ int wbase[WPB];
    int i = blockIdx.x * 256 + threadIdx.x;
    int lane = threadIdx.x & 31, wave = threadIdx.x >> 5;
    int v = (i < M_ && q[i] != 0) ? 1 : 0;
    unsigned lm = (unsigned)__ballot(v);
    if (lane == 0) wbase[wave] = __popc(lm);
    __syncthreads();
    if (threadIdx.x == 0) {
        int s = coff[blockIdx.x];
        for (int k = 0; k < WPB; ++k) { int t = wbase[k]; wbase[k] = s; s += t; }
    }
    __syncthreads();
    if (v) {
        int rank = wbase[wave] + __popc(lm & ((1u << lane) - 1u));
        if (rank < NQ_) qi[rank] = i;
    }
}

__global__ void __launch_bounds__(256) gather_rows_kernel(const float* __restrict__ src,
                                                          const int* __restrict__ idx,
                                                          float* __restrict__ dst, int nrows) {
    int t = blockIdx.x * blockDim.x + threadIdx.x;
    if (t < nrows * 64) {
        int r = t >> 6, c = t & 63;
        dst[t] = src[(size_t)idx[r] * 64 + c];
    }
}

__global__ void __launch_bounds__(256) degree_kernel(const int* __restrict__ ht,
                                                     float* __restrict__ cnt, int M_) {
    int i = blockIdx.x * blockDim.x + threadIdx.x;
    if (i < M_) {
        atomicAdd(&cnt[ht[2 * i + 1]], 1.f);  // tails
        atomicAdd(&cnt[ht[2 * i + 0]], 1.f);  // heads
    }
}

// ---------------------------------------------------------------------------
// Message passing: mf = [H[head]|E] @ Wf + bf  -> atomicAdd into agg[tail]
//                  mb = [H[tail]|E] @ Wb + bb  -> atomicAdd into agg[head]
// ---------------------------------------------------------------------------
__global__ void __launch_bounds__(BLOCK) message_kernel(
    const float* __restrict__ H, const float* __restrict__ E, const int* __restrict__ ht,
    const v16h* __restrict__ bfragF, const v16h* __restrict__ bfragB,
    const float* __restrict__ biasF, const float* __restrict__ biasB,
    float* __restrict__ agg, int nedges) {
    __shared__ float lds[WPB][16 * 64];
    __shared__ int   sdst[WPB][16];
    const int lane = threadIdx.x & 31, wave = threadIdx.x >> 5;
    const int tile = blockIdx.x * WPB + wave;
    if (tile * 16 >= nedges) return;
    const int e = tile * 16 + (lane & 15);
    const int head = ht[2 * e], tail = ht[2 * e + 1];
    float* s = lds[wave];

    {   // forward messages -> tails
        const float* srcs[2] = { H + (size_t)head * 64, E + (size_t)e * 64 };
        v8f c[4] = {};
        wmma_chunks<2>(srcs, bfragF, lane, c);
        store_c_lds(s, c, lane);
        if (lane < 16) sdst[wave][lane] = tail;
        for (int idx = lane; idx < 1024; idx += 32) {
            int rr = idx >> 6, cc = idx & 63;
            atomicAdd(&agg[(size_t)sdst[wave][rr] * 64 + cc], s[idx] + biasF[cc]);
        }
    }
    {   // backward messages -> heads
        const float* srcs[2] = { H + (size_t)tail * 64, E + (size_t)e * 64 };
        v8f c[4] = {};
        wmma_chunks<2>(srcs, bfragB, lane, c);
        store_c_lds(s, c, lane);
        if (lane < 16) sdst[wave][lane] = head;
        for (int idx = lane; idx < 1024; idx += 32) {
            int rr = idx >> 6, cc = idx & 63;
            atomicAdd(&agg[(size_t)sdst[wave][rr] * 64 + cc], s[idx] + biasB[cc]);
        }
    }
}

// H = LN(leaky(agg / cnt) + H)
__global__ void __launch_bounds__(256) node_update_kernel(
    float* __restrict__ H, const float* __restrict__ agg, const float* __restrict__ cnt,
    const float* __restrict__ g, const float* __restrict__ be, int Nn) {
    int row = blockIdx.x * blockDim.x + threadIdx.x;
    if (row >= Nn) return;
    const float* a = agg + (size_t)row * 64;
    float* h = H + (size_t)row * 64;
    float cv = cnt[row];
    float ic = cv > 0.f ? 1.f / cv : 0.f;
    float mu = 0.f;
    for (int k = 0; k < 64; ++k) mu += leakyf(a[k] * ic) + h[k];
    mu *= (1.f / 64.f);
    float var = 0.f;
    for (int k = 0; k < 64; ++k) { float d = leakyf(a[k] * ic) + h[k] - mu; var += d * d; }
    float inv = rsqrtf(var * (1.f / 64.f) + 1e-5f);
    for (int k = 0; k < 64; ++k) {
        float x = leakyf(a[k] * ic) + h[k];
        h[k] = (x - mu) * inv * g[k] + be[k];
    }
}

// E = LN(leaky([H[head]|E|H[tail]] @ eu_W + eu_b) + E)   (in-place per row)
__global__ void __launch_bounds__(BLOCK) edge_update_kernel(
    const float* __restrict__ H, float* __restrict__ E, const int* __restrict__ ht,
    const v16h* __restrict__ bfrag, const float* __restrict__ bias,
    const float* __restrict__ g, const float* __restrict__ be, int nedges) {
    __shared__ float lds[WPB][16 * 64];
    const int lane = threadIdx.x & 31, wave = threadIdx.x >> 5;
    const int tile = blockIdx.x * WPB + wave;
    if (tile * 16 >= nedges) return;
    const int e = tile * 16 + (lane & 15);
    const int head = ht[2 * e], tail = ht[2 * e + 1];
    const float* srcs[3] = { H + (size_t)head * 64, E + (size_t)e * 64, H + (size_t)tail * 64 };
    v8f c[4] = {};
    wmma_chunks<3>(srcs, bfrag, lane, c);
    float* s = lds[wave];
    store_c_lds(s, c, lane);
    if (lane < 16) {
        float* row = s + lane * 64;
        float* erow = E + (size_t)(tile * 16 + lane) * 64;
        float mu = 0.f;
        for (int k = 0; k < 64; ++k) { float x = leakyf(row[k] + bias[k]) + erow[k]; row[k] = x; mu += x; }
        mu *= (1.f / 64.f);
        float var = 0.f;
        for (int k = 0; k < 64; ++k) { float d = row[k] - mu; var += d * d; }
        float inv = rsqrtf(var * (1.f / 64.f) + 1e-5f);
        for (int k = 0; k < 64; ++k)
            erow[k] = (row[k] - mu) * inv * g[k] + be[k];
    }
}

// out = leaky([E[q]|E0[q]|H[h]|H0[h]|H[t]|H0[t]] @ W1 + b1) @ W2 + b2
__global__ void __launch_bounds__(BLOCK) classifier_kernel(
    const float* __restrict__ E, const float* __restrict__ E0q,
    const float* __restrict__ H, const float* __restrict__ H0,
    const int* __restrict__ qi, const int* __restrict__ ht,
    const v16h* __restrict__ bfrag, const float* __restrict__ b1,
    const float* __restrict__ W2, const float* __restrict__ b2,
    float* __restrict__ out, int nq) {
    __shared__ float lds[WPB][16 * 64];
    const int lane = threadIdx.x & 31, wave = threadIdx.x >> 5;
    const int tile = blockIdx.x * WPB + wave;
    if (tile * 16 >= nq) return;
    const int qpos = tile * 16 + (lane & 15);
    const int q = qi[qpos];
    const int head = ht[2 * q], tail = ht[2 * q + 1];
    const float* srcs[6] = { E  + (size_t)q * 64,    E0q + (size_t)qpos * 64,
                             H  + (size_t)head * 64, H0  + (size_t)head * 64,
                             H  + (size_t)tail * 64, H0  + (size_t)tail * 64 };
    v8f c[4] = {};
    wmma_chunks<6>(srcs, bfrag, lane, c);
    float* s = lds[wave];
    store_c_lds(s, c, lane);
    if (lane < 16) {
        const float* row = s + lane * 64;
        float acc = 0.f;
        for (int k = 0; k < 64; ++k) acc += leakyf(row[k] + b1[k]) * W2[k];
        out[tile * 16 + lane] = acc + b2[0];
    }
}

// ---------------------------------------------------------------------------
extern "C" void kernel_launch(void* const* d_in, const int* in_sizes, int n_in,
                              void* d_out, int out_size, void* d_ws, size_t ws_size,
                              hipStream_t stream) {
    (void)in_sizes; (void)n_in; (void)out_size; (void)ws_size;
    const int N = 50000, M = 800000, L = 2, NQ = 2048;

    const int*   ht          = (const int*)  d_in[0];
    const int*   r_tensor    = (const int*)  d_in[1];
    const float* entity_feat = (const float*)d_in[2];
    const int*   queries     = (const int*)  d_in[3];
    const float* rel_table   = (const float*)d_in[4];
    const float* W_ent    = (const float*)d_in[5];
    const float* b_ent    = (const float*)d_in[6];
    const float* g_ent    = (const float*)d_in[7];
    const float* be_ent   = (const float*)d_in[8];
    const float* W_edge   = (const float*)d_in[9];
    const float* b_edge   = (const float*)d_in[10];
    const float* g_edge   = (const float*)d_in[11];
    const float* be_edge  = (const float*)d_in[12];
    const float* mp_Wf    = (const float*)d_in[13];
    const float* mp_bf    = (const float*)d_in[14];
    const float* mp_Wb    = (const float*)d_in[15];
    const float* mp_bb    = (const float*)d_in[16];
    const float* mp_g     = (const float*)d_in[17];
    const float* mp_be    = (const float*)d_in[18];
    const float* eu_W     = (const float*)d_in[19];
    const float* eu_b     = (const float*)d_in[20];
    const float* eu_g     = (const float*)d_in[21];
    const float* eu_be    = (const float*)d_in[22];
    const float* cls_W1   = (const float*)d_in[23];
    const float* cls_b1   = (const float*)d_in[24];
    const float* cls_W2   = (const float*)d_in[25];
    const float* cls_b2   = (const float*)d_in[26];
    float* out = (float*)d_out;

    char* ws = (char*)d_ws;
    size_t off = 0;
    auto alloc = [&](size_t bytes) -> void* {
        off = (off + 255) & ~(size_t)255;
        void* p = ws + off;
        off += bytes;
        return p;
    };

    float* H0   = (float*)alloc((size_t)N * 64 * 4);
    float* H    = (float*)alloc((size_t)N * 64 * 4);
    float* agg  = (float*)alloc((size_t)N * 64 * 4);
    float* E    = (float*)alloc((size_t)M * 64 * 4);
    float* E0q  = (float*)alloc((size_t)NQ * 64 * 4);
    float* cnt  = (float*)alloc((size_t)N * 4);
    int*   qi   = (int*)  alloc((size_t)NQ * 4);
    const int nch = (M + 255) / 256;
    int* ccnt = (int*)alloc((size_t)nch * 4);
    int* coff = (int*)alloc((size_t)nch * 4);
    _Float16* fEnt  = (_Float16*)alloc((size_t)768 * 64 * 2);
    _Float16* fEdge = (_Float16*)alloc((size_t)768 * 64 * 2);
    _Float16* fWf   = (_Float16*)alloc((size_t)L * 128 * 64 * 2);
    _Float16* fWb   = (_Float16*)alloc((size_t)L * 128 * 64 * 2);
    _Float16* fEu   = (_Float16*)alloc((size_t)L * 192 * 64 * 2);
    _Float16* fC1   = (_Float16*)alloc((size_t)384 * 64 * 2);

    // 1) pack weights into WMMA fragment order (f16)
    pack_b_kernel<<<(768 * 64 + 255) / 256, 256, 0, stream>>>(W_ent, fEnt, 768);
    pack_b_kernel<<<(768 * 64 + 255) / 256, 256, 0, stream>>>(W_edge, fEdge, 768);
    for (int l = 0; l < L; ++l) {
        pack_b_kernel<<<(128 * 64 + 255) / 256, 256, 0, stream>>>(mp_Wf + (size_t)l * 128 * 64, fWf + (size_t)l * 128 * 64, 128);
        pack_b_kernel<<<(128 * 64 + 255) / 256, 256, 0, stream>>>(mp_Wb + (size_t)l * 128 * 64, fWb + (size_t)l * 128 * 64, 128);
        pack_b_kernel<<<(192 * 64 + 255) / 256, 256, 0, stream>>>(eu_W + (size_t)l * 192 * 64, fEu + (size_t)l * 192 * 64, 192);
    }
    pack_b_kernel<<<(384 * 64 + 255) / 256, 256, 0, stream>>>(cls_W1, fC1, 384);

    // 2) query compaction
    count_chunks_kernel<<<nch, 256, 0, stream>>>(queries, ccnt, M);
    scan_chunks_kernel<<<1, 1, 0, stream>>>(ccnt, coff, nch);
    write_qi_kernel<<<nch, 256, 0, stream>>>(queries, coff, qi, M, NQ);

    // 3) projections
    node_proj_kernel<<<(N / 16 + WPB - 1) / WPB, BLOCK, 0, stream>>>(
        entity_feat, (const v16h*)fEnt, b_ent, g_ent, be_ent, H0, H, N);
    edge_proj_kernel<<<(M / 16 + WPB - 1) / WPB, BLOCK, 0, stream>>>(
        rel_table, r_tensor, (const v16h*)fEdge, b_edge, g_edge, be_edge, E, M);
    gather_rows_kernel<<<(NQ * 64 + 255) / 256, 256, 0, stream>>>(E, qi, E0q, NQ);  // E0[qi]

    // 4) degrees
    zero_kernel<<<(N + 255) / 256, 256, 0, stream>>>(cnt, N);
    degree_kernel<<<(M + 255) / 256, 256, 0, stream>>>(ht, cnt, M);

    // 5) message-passing layers
    for (int l = 0; l < L; ++l) {
        zero_kernel<<<(N * 64 + 255) / 256, 256, 0, stream>>>(agg, N * 64);
        message_kernel<<<(M / 16 + WPB - 1) / WPB, BLOCK, 0, stream>>>(
            H, E, ht,
            (const v16h*)(fWf + (size_t)l * 128 * 64),
            (const v16h*)(fWb + (size_t)l * 128 * 64),
            mp_bf + (size_t)l * 64, mp_bb + (size_t)l * 64, agg, M);
        node_update_kernel<<<(N + 255) / 256, 256, 0, stream>>>(
            H, agg, cnt, mp_g + (size_t)l * 64, mp_be + (size_t)l * 64, N);
        edge_update_kernel<<<(M / 16 + WPB - 1) / WPB, BLOCK, 0, stream>>>(
            H, E, ht, (const v16h*)(fEu + (size_t)l * 192 * 64),
            eu_b + (size_t)l * 64, eu_g + (size_t)l * 64, eu_be + (size_t)l * 64, M);
    }

    // 6) classifier head
    classifier_kernel<<<(NQ / 16 + WPB - 1) / WPB, BLOCK, 0, stream>>>(
        E, E0q, H, H0, qi, ht, (const v16h*)fC1, cls_b1, cls_W2, cls_b2, out, NQ);
}
